// FourierEncoder_7069516169694
// MI455X (gfx1250) — matrix operand order repaired and checked by reading, not compiled, not run on hardware
//
#include <hip/hip_runtime.h>

// ---------------------------------------------------------------------------
// STFT-as-GEMM for MI455X (gfx1250): out[b,m,f] = sum_k basis[m,k]*x[b,f*256+k]
// basis precomputed into WMMA A-fragment layout (bf16), frames staged in LDS.
// Round 2: 2x2 register blocking per wave (2 mtiles x 2 ntiles) -> 2 loads/WMMA.
// ---------------------------------------------------------------------------

typedef __bf16 bf16_t;
typedef __attribute__((ext_vector_type(16))) __bf16 v16bf;
typedef __attribute__((ext_vector_type(8)))  __bf16 v8bf;
typedef __attribute__((ext_vector_type(8)))  float  v8f;
typedef __attribute__((ext_vector_type(4)))  float  v4f;

#define T_LEN     524288      // samples per batch
#define BATCH     16
#define CH        1026        // 513 real + 513 imag output channels
#define NBINS     513
#define KSIZE     1024
#define STRIDE    256
#define NFRAMES   2045        // (524288-1024)/256 + 1
#define KT_COUNT  32          // 1024 / 32 (bf16 WMMA K)
#define MT_PER_BLK 8          // channel tiles (of 16) per block -> 128 ch
#define MBLOCKS   9           // ceil(1026/128) -> 72 mtiles padded
#define MT_TOTAL  (MBLOCKS * MT_PER_BLK)   // 72
#define NBLK      32          // frames per block (2 N-tiles of 16)
#define NBLOCKS   64          // ceil(2045/32)
#define LDS_ROW   1032        // 1024 + 8 halves pad -> conflict-free b128 reads
#define FRAG_HALVES 512       // 32 lanes * 16 halves per (mtile,ktile) fragment

// ---------------------------------------------------------------------------
// Kernel 1: build windowed Fourier basis directly in WMMA A-fragment layout.
// A-matrix 16x32 bf16 layout (ISA 7.12.2): lane = g*16 + (m%16);
//   VGPR v, pair p: g==0,v<4 -> k=2v+p ; g==1,v<4 -> 8+2v+p ;
//                   g==0,v>=4 -> 16+2(v-4)+p ; g==1,v>=4 -> 24+2(v-4)+p
// frag element index: ((mt*32 + kt)*32 + lane)*16 + h
// ---------------------------------------------------------------------------
__global__ __launch_bounds__(256)
void fourier_basis_frag_kernel(const float* __restrict__ freq,
                               const float* __restrict__ window,
                               bf16_t* __restrict__ frag) {
    const int e    = blockIdx.x * 256 + threadIdx.x;
    const int fi   = e >> 9;            // fragment index = mt*32 + kt
    const int q    = e & 511;
    const int lane = q >> 4;
    const int h    = q & 15;
    const int mt   = fi / KT_COUNT;
    const int kt   = fi - mt * KT_COUNT;
    const int g    = lane >> 4;
    const int m    = mt * 16 + (lane & 15);
    const int v    = h >> 1;
    const int p    = h & 1;
    const int kloc = (v < 4 ? 2 * v + p : 16 + 2 * (v - 4) + p) + g * 8;
    const int k    = kt * 32 + kloc;

    float val = 0.0f;
    if (m < NBINS) {                       // real part: cos(-w*k) = cos(w*k)
        val = cosf(freq[m] * (float)k) * window[k];
    } else if (m < CH) {                   // imag part: sin(-w*k) = -sin(w*k)
        val = -sinf(freq[m - NBINS] * (float)k) * window[k];
    }
    frag[e] = (bf16_t)val;
}

// ---------------------------------------------------------------------------
// Kernel 2: block = 4 waves (128 threads); block tile = 128 ch x 32 frames.
// Wave w owns mtiles {mb*8+2w, mb*8+2w+1} x both N-tiles -> 2x2 accumulators.
// B fragment (K=32 x N=16 bf16): lanes 0-15 hold k=0..7,16..23 for n=lane,
// lanes 16-31 hold k=8..15,24..31 -> per lane two 16-byte LDS reads.
// ---------------------------------------------------------------------------
__global__ __launch_bounds__(128)
void stft_wmma_kernel(const float* __restrict__ x,
                      const bf16_t* __restrict__ frag,
                      float* __restrict__ out) {
    __shared__ bf16_t xs[NBLK][LDS_ROW];   // 66,048 B

    const int b  = blockIdx.z;
    const int mb = blockIdx.y;
    const int nb = blockIdx.x;
    const int tid = threadIdx.x;
    const float* __restrict__ xb = x + (size_t)b * T_LEN;

    // ---- stage 32 overlapping frames -> bf16 LDS (each thread: 256 samples)
    {
        const int row = tid >> 2;                  // frame within block
        const int k0  = (tid & 3) << 8;            // 256-sample chunk
        const long g0 = (long)nb * (NBLK * STRIDE) + (long)row * STRIDE + k0;
        bf16_t* dst = &xs[row][k0];
        if (g0 + 256 <= T_LEN) {
            #pragma unroll
            for (int j = 0; j < 256; j += 4) {
                v4f f = *(const v4f*)(xb + g0 + j);
                dst[j + 0] = (bf16_t)f[0];
                dst[j + 1] = (bf16_t)f[1];
                dst[j + 2] = (bf16_t)f[2];
                dst[j + 3] = (bf16_t)f[3];
            }
        } else {                                    // tail block: guarded
            for (int j = 0; j < 256; ++j) {
                const long gi = g0 + j;
                dst[j] = (bf16_t)((gi < T_LEN) ? xb[gi] : 0.0f);
            }
        }
    }
    __syncthreads();

    // ---- WMMA main loop: 2 mtiles x 2 ntiles per wave
    const int wave = tid >> 5;
    const int lane = tid & 31;
    const int g    = lane >> 4;
    const int l    = lane & 15;
    const int mt0  = mb * MT_PER_BLK + wave * 2;
    const int mt1  = mt0 + 1;

    const bf16_t* __restrict__ arow0 =
        frag + ((size_t)mt0 * KT_COUNT * 32 + lane) * 16;  // += 512 per kt
    const bf16_t* __restrict__ arow1 =
        frag + ((size_t)mt1 * KT_COUNT * 32 + lane) * 16;
    const bf16_t* __restrict__ r0 = &xs[l][g * 8];         // N-tile 0
    const bf16_t* __restrict__ r1 = &xs[l + 16][g * 8];    // N-tile 1

    v8f acc00 = {};   // (mt0, nt0)
    v8f acc01 = {};   // (mt0, nt1)
    v8f acc10 = {};   // (mt1, nt0)
    v8f acc11 = {};   // (mt1, nt1)

    for (int kt = 0; kt < KT_COUNT; ++kt) {
        v16bf a0 = *(const v16bf*)(arow0 + (size_t)kt * FRAG_HALVES);
        v16bf a1 = *(const v16bf*)(arow1 + (size_t)kt * FRAG_HALVES);
        if (kt + 1 < KT_COUNT) { // gfx1250 global_prefetch for next A fragments
            __builtin_prefetch(arow0 + (size_t)(kt + 1) * FRAG_HALVES, 0, 3);
            __builtin_prefetch(arow1 + (size_t)(kt + 1) * FRAG_HALVES, 0, 3);
        }

        v8bf b0lo = *(const v8bf*)(r0 + kt * 32);
        v8bf b0hi = *(const v8bf*)(r0 + kt * 32 + 16);
        v8bf b1lo = *(const v8bf*)(r1 + kt * 32);
        v8bf b1hi = *(const v8bf*)(r1 + kt * 32 + 16);
        v16bf b0 = __builtin_shufflevector(b0lo, b0hi,
            0, 1, 2, 3, 4, 5, 6, 7, 8, 9, 10, 11, 12, 13, 14, 15);
        v16bf b1 = __builtin_shufflevector(b1lo, b1hi,
            0, 1, 2, 3, 4, 5, 6, 7, 8, 9, 10, 11, 12, 13, 14, 15);

        acc00 = __builtin_amdgcn_wmma_f32_16x16x32_bf16(
                    false, a0, false, b0, (short)0, acc00, false, false);
        acc01 = __builtin_amdgcn_wmma_f32_16x16x32_bf16(
                    false, a0, false, b1, (short)0, acc01, false, false);
        acc10 = __builtin_amdgcn_wmma_f32_16x16x32_bf16(
                    false, a1, false, b0, (short)0, acc10, false, false);
        acc11 = __builtin_amdgcn_wmma_f32_16x16x32_bf16(
                    false, a1, false, b1, (short)0, acc11, false, false);
    }

    // ---- store: C/D layout -> VGPR r holds M = mt*16 + g*8 + r, N = l
    const int f0 = nb * NBLK + l;
    const int f1 = f0 + 16;
    float* __restrict__ ob = out + (size_t)b * CH * NFRAMES;

    const int mbase0 = mt0 * 16 + g * 8;
    #pragma unroll
    for (int r = 0; r < 8; ++r) {
        const int m = mbase0 + r;
        if (m < CH) {
            float* orow = ob + (size_t)m * NFRAMES;
            if (f0 < NFRAMES) orow[f0] = acc00[r];
            if (f1 < NFRAMES) orow[f1] = acc01[r];
        }
    }
    const int mbase1 = mt1 * 16 + g * 8;
    #pragma unroll
    for (int r = 0; r < 8; ++r) {
        const int m = mbase1 + r;
        if (m < CH) {
            float* orow = ob + (size_t)m * NFRAMES;
            if (f0 < NFRAMES) orow[f0] = acc10[r];
            if (f1 < NFRAMES) orow[f1] = acc11[r];
        }
    }
}

// ---------------------------------------------------------------------------
extern "C" void kernel_launch(void* const* d_in, const int* in_sizes, int n_in,
                              void* d_out, int out_size, void* d_ws, size_t ws_size,
                              hipStream_t stream) {
    const float* x      = (const float*)d_in[0];   // (16,1,524288)
    const float* freq   = (const float*)d_in[1];   // (513,)
    const float* window = (const float*)d_in[2];   // (1024,)
    float* out          = (float*)d_out;           // (16,1026,2045)
    bf16_t* frag        = (bf16_t*)d_ws;           // 72*32*512*2 = 2,359,296 B

    const int total_frag_elems = MT_TOTAL * KT_COUNT * FRAG_HALVES;
    fourier_basis_frag_kernel<<<total_frag_elems / 256, 256, 0, stream>>>(
        freq, window, frag);

    dim3 grid(NBLOCKS, MBLOCKS, BATCH);
    stft_wmma_kernel<<<grid, 128, 0, stream>>>(x, frag, out);
}
